// LSTMCell_52338471469772
// MI455X (gfx1250) — matrix operand order repaired and checked by reading, not compiled, not run on hardware
//
#include <hip/hip_runtime.h>

// CDNA5 / gfx1250 LSTM cell: fused dual-GEMM (x- and h-projections, 4 gates)
// via v_wmma_f32_16x16x32_bf16, double-buffered LDS staging with f32->bf16
// conversion in flight, branch-free fast activations in the fused epilogue.

typedef __attribute__((ext_vector_type(16))) __bf16 v16bf;
typedef __attribute__((ext_vector_type(8)))  float  v8f;

#define B_DIM 8192
#define IN_D  512
#define HID   512
#define KTOT  1024            // concat(x, h) reduction dim
#define BM    128             // batch rows per block
#define BH    32              // hidden cols per block (per gate)
#define KT    32              // K per WMMA step
#define NIT   (KTOT / KT)     // 32 K-iterations
#define LDSTR 40              // padded LDS row stride (bf16 elems, 80B)

struct Params {
    const float *x, *h, *c;
    const float *Wfx, *Wfh, *Wgx, *Wgh, *Wix, *Wih, *Wox, *Woh;
    const float *bfx, *bfh, *bgx, *bgh, *bix, *bih, *box_, *boh;
    float *hOut, *cOut;
};

union Frag { v16bf v; float4 q[2]; };

__device__ __forceinline__ void cvt_store8(const float4 a, const float4 b, __bf16* dst) {
    union { __bf16 e[8]; float4 q; } u;
    u.e[0] = (__bf16)a.x; u.e[1] = (__bf16)a.y; u.e[2] = (__bf16)a.z; u.e[3] = (__bf16)a.w;
    u.e[4] = (__bf16)b.x; u.e[5] = (__bf16)b.y; u.e[6] = (__bf16)b.z; u.e[7] = (__bf16)b.w;
    *reinterpret_cast<float4*>(dst) = u.q;   // ds_store_b128
}

// Branch-free activations: native v_exp_f32 + v_rcp_f32, no ocml call tree.
__device__ __forceinline__ float fast_rcp(float x) { return __builtin_amdgcn_rcpf(x); }
__device__ __forceinline__ float fast_sigmoid(float x) {
    return fast_rcp(1.0f + __expf(-x));
}
__device__ __forceinline__ float fast_tanh(float x) {
    // tanh(x) = 1 - 2/(e^{2x}+1); saturates cleanly to +/-1 at the extremes.
    return 1.0f - 2.0f * fast_rcp(__expf(2.0f * x) + 1.0f);
}

__launch_bounds__(256, 2)
__global__ void lstm_wmma_kernel(Params p) {
    __shared__ alignas(16) __bf16 sA[2][BM][LDSTR];       // 2 x 128 x 32 (padded)
    __shared__ alignas(16) __bf16 sB[2][4][BH][LDSTR];    // 2 x 4 gates x 32 x 32

    const int tid  = threadIdx.x;
    const int lane = tid & 31;
    const int wave = tid >> 5;
    const int half = lane >> 4;       // 0: lanes 0-15, 1: lanes 16-31
    const int l15  = lane & 15;

    const int hBase = blockIdx.x * BH;     // hidden-unit tile base
    const int mBase = blockIdx.y * BM;     // batch tile base

    // ---- per-thread staging indices (constant across K-loop) ----
    const int aRow = tid >> 2;                 // 0..63  (and +64 for 2nd chunk)
    const int cOff = (tid & 3) * 8;            // 0,8,16,24 (K sub-offset)
    const int bN   = (tid & 127) >> 2;         // 0..31 (hidden col in tile)
    const int g0   = tid >> 7;                 // gate 0/1, second store g0+2
    const size_t aOff0 = (size_t)(mBase + aRow) * IN_D + cOff;
    const size_t aOff1 = (size_t)(mBase + aRow + 64) * IN_D + cOff;
    const size_t wOff  = (size_t)(hBase + bN) * IN_D + cOff;

    v8f acc[4][2];
    #pragma unroll
    for (int g = 0; g < 4; ++g) { acc[g][0] = {}; acc[g][1] = {}; }

    float4 stg[8];

    // ---- global fetch of one K-tile into registers ----
    auto global_fetch = [&](int k0) {
        const bool  isX  = (k0 < IN_D);
        const int   kOff = isX ? k0 : (k0 - IN_D);
        const float* aP = (isX ? p.x : p.h) + kOff;
        const float* w0 = isX ? (g0 ? p.Wgx : p.Wfx) : (g0 ? p.Wgh : p.Wfh);
        const float* w1 = isX ? (g0 ? p.Wox : p.Wix) : (g0 ? p.Woh : p.Wih);
        const float4* s;
        s = reinterpret_cast<const float4*>(aP + aOff0);        stg[0] = s[0]; stg[1] = s[1];
        s = reinterpret_cast<const float4*>(aP + aOff1);        stg[2] = s[0]; stg[3] = s[1];
        s = reinterpret_cast<const float4*>(w0 + kOff + wOff);  stg[4] = s[0]; stg[5] = s[1];
        s = reinterpret_cast<const float4*>(w1 + kOff + wOff);  stg[6] = s[0]; stg[7] = s[1];
    };

    // ---- convert + store staged registers into LDS buffer `buf` ----
    auto stage_store = [&](int buf) {
        cvt_store8(stg[0], stg[1], &sA[buf][aRow][cOff]);
        cvt_store8(stg[2], stg[3], &sA[buf][aRow + 64][cOff]);
        cvt_store8(stg[4], stg[5], &sB[buf][g0][bN][cOff]);
        cvt_store8(stg[6], stg[7], &sB[buf][g0 + 2][bN][cOff]);
    };

    // ---- prologue: stage tile 0 ----
    global_fetch(0);
    stage_store(0);
    __syncthreads();

    for (int it = 0; it < NIT; ++it) {
        const int  buf     = it & 1;
        const bool hasNext = (it + 1) < NIT;

        // issue next tile's global loads early (overlap with WMMA below)
        if (hasNext) global_fetch((it + 1) * KT);

        // ---- A fragment: lane=row l15, K contiguous 16*half..16*half+15 ----
        Frag af;
        {
            const __bf16* aP = &sA[buf][wave * 16 + l15][half * 16];
            af.q[0] = *reinterpret_cast<const float4*>(aP);
            af.q[1] = *reinterpret_cast<const float4*>(aP + 8);
        }
        // ---- all 8 B fragments first (staged s_wait_dscnt), then 8 WMMAs ----
        Frag bfr[8];
        #pragma unroll
        for (int g = 0; g < 4; ++g) {
            #pragma unroll
            for (int t = 0; t < 2; ++t) {
                const __bf16* bP = &sB[buf][g][t * 16 + l15][half * 16];
                bfr[g * 2 + t].q[0] = *reinterpret_cast<const float4*>(bP);
                bfr[g * 2 + t].q[1] = *reinterpret_cast<const float4*>(bP + 8);
            }
        }
        #pragma unroll
        for (int g = 0; g < 4; ++g) {
            #pragma unroll
            for (int t = 0; t < 2; ++t) {
                acc[g][t] = __builtin_amdgcn_wmma_f32_16x16x32_bf16(
                    false, af.v, false, bfr[g * 2 + t].v, (short)0, acc[g][t],
                    false, false);
            }
        }

        // stage next tile into the other buffer, then one barrier per iter
        if (hasNext) stage_store(buf ^ 1);
        __syncthreads();
    }

    // ---- fused epilogue: bias + gates + cell update ----
    // C/D layout: VGPR v, lane l -> M = v + 8*(l>>4), N = l&15
    const int rWave = mBase + wave * 16 + half * 8;
    #pragma unroll
    for (int t = 0; t < 2; ++t) {
        const int col = hBase + t * 16 + l15;
        const float bF = p.bfx[col] + p.bfh[col];
        const float bG = p.bgx[col] + p.bgh[col];
        const float bI = p.bix[col] + p.bih[col];
        const float bO = p.box_[col] + p.boh[col];
        #pragma unroll
        for (int v = 0; v < 8; ++v) {
            const int row = rWave + v;
            const float f_t = fast_sigmoid(acc[0][t][v] + bF);
            const float g_t = fast_tanh   (acc[1][t][v] + bG);
            const float i_t = fast_sigmoid(acc[2][t][v] + bI);
            const float o_t = fast_sigmoid(acc[3][t][v] + bO);
            const float cPrev = p.c[(size_t)row * HID + col];
            const float cNew  = f_t * cPrev + i_t * g_t;
            const float hNew  = o_t * fast_tanh(cNew);
            p.cOut[(size_t)row * HID + col] = cNew;
            p.hOut[(size_t)row * HID + col] = hNew;
        }
    }
}

extern "C" void kernel_launch(void* const* d_in, const int* in_sizes, int n_in,
                              void* d_out, int out_size, void* d_ws, size_t ws_size,
                              hipStream_t stream) {
    (void)in_sizes; (void)n_in; (void)out_size; (void)d_ws; (void)ws_size;
    Params p;
    p.x    = (const float*)d_in[0];
    p.h    = (const float*)d_in[1];
    p.c    = (const float*)d_in[2];
    p.Wfx  = (const float*)d_in[3];   p.bfx = (const float*)d_in[4];
    p.Wfh  = (const float*)d_in[5];   p.bfh = (const float*)d_in[6];
    p.Wix  = (const float*)d_in[7];   p.bix = (const float*)d_in[8];
    p.Wih  = (const float*)d_in[9];   p.bih = (const float*)d_in[10];
    p.Wgx  = (const float*)d_in[11];  p.bgx = (const float*)d_in[12];
    p.Wgh  = (const float*)d_in[13];  p.bgh = (const float*)d_in[14];
    p.Wox  = (const float*)d_in[15];  p.box_ = (const float*)d_in[16];
    p.Woh  = (const float*)d_in[17];  p.boh = (const float*)d_in[18];
    p.hOut = (float*)d_out;
    p.cOut = (float*)d_out + (size_t)B_DIM * HID;

    dim3 grid(HID / BH, B_DIM / BM);   // 16 x 64 blocks
    dim3 block(256);                   // 8 wave32s
    lstm_wmma_kernel<<<grid, block, 0, stream>>>(p);
}
